// MultiHeadAttention_57217554317896
// MI455X (gfx1250) — matrix operand order repaired and checked by reading, not compiled, not run on hardware
//
#include <hip/hip_runtime.h>
#include <hip/hip_bf16.h>

#define NUM_HEADS 8
#define HEAD_SIZE 16
#define N_EMBD    128
#define BATCH     2
#define SEQ       4096
#define MTILES    4              // 64 query rows per wave in flash attention

typedef __attribute__((ext_vector_type(16))) __bf16       v16bf;
typedef __attribute__((ext_vector_type(8)))  float        v8f;
typedef __attribute__((ext_vector_type(8)))  unsigned int v8u;

static __device__ __forceinline__ unsigned short f2bf(float f) {
    unsigned int u = __builtin_bit_cast(unsigned int, f);
    u += 0x7FFFu + ((u >> 16) & 1u);           // round-to-nearest-even
    return (unsigned short)(u >> 16);
}

// ---------------------------------------------------------------- conversion
__global__ void cvt_bf16_kernel(const float* __restrict__ src,
                                unsigned short* __restrict__ dst, int n) {
    int i = blockIdx.x * blockDim.x + threadIdx.x;
    for (; i < n; i += gridDim.x * blockDim.x) dst[i] = f2bf(src[i]);
}

// ------------------------------------------------- wave-level 16x16xC GEMM
// A: 16 rows (stride sA) of bf16, K = N_EMBD.  W: out-feature rows (B = W^T).
static __device__ __forceinline__ v8f gemm16x16xC(
    const unsigned short* __restrict__ A, int sA,
    const unsigned short* __restrict__ W, int sW, int lane)
{
    const int half = lane >> 4;
    const int l15  = lane & 15;
    v8f c = {0.f,0.f,0.f,0.f,0.f,0.f,0.f,0.f};
    const unsigned short* arow = A + (size_t)l15 * sA;
    const unsigned short* wrow = W + (size_t)l15 * sW;
#pragma unroll
    for (int kk = 0; kk < N_EMBD; kk += 32) {
        v8u au, bu;
#pragma unroll
        for (int j = 0; j < 4; ++j)   // A VGPR 0-3: K = kk + half*16 + {0..7}
            au[j] = *(const unsigned int*)(arow + kk + half*16 + 2*j);
#pragma unroll
        for (int j = 4; j < 8; ++j)   // A VGPR 4-7: K = kk + half*16 + 8 + ..
            au[j] = *(const unsigned int*)(arow + kk + half*16 + 2*j);
#pragma unroll
        for (int j = 0; j < 8; ++j)   // B: lane half selects K 0-15 / 16-31
            bu[j] = *(const unsigned int*)(wrow + kk + half*16 + 2*j);
        v16bf a = __builtin_bit_cast(v16bf, au);
        v16bf b = __builtin_bit_cast(v16bf, bu);
        c = __builtin_amdgcn_wmma_f32_16x16x32_bf16(false, a, false, b,
                                                    (short)0, c, false, false);
    }
    return c;
}

// ---------------------------------------------------------------- QKV GEMM
__global__ void qkv_gemm_kernel(const unsigned short* __restrict__ xb,
                                const unsigned short* __restrict__ wq,
                                const unsigned short* __restrict__ wk,
                                const unsigned short* __restrict__ wv,
                                unsigned short* __restrict__ Q,
                                unsigned short* __restrict__ Kd,
                                unsigned short* __restrict__ V)
{
    const int lane  = threadIdx.x;
    const int mtile = blockIdx.x;              // (B*T)/16 tiles
    const int h     = blockIdx.y;              // head == 16-wide N tile
    const int op    = blockIdx.z;              // 0=Q 1=K 2=V
    const unsigned short* W = (op == 0) ? wq : (op == 1) ? wk : wv;
    v8f c = gemm16x16xC(xb + (size_t)mtile * 16 * N_EMBD, N_EMBD,
                        W + (size_t)h * 16 * N_EMBD, N_EMBD, lane);
    const float scale = (op == 0) ? 0.25f : 1.0f;     // D^-1/2 folded into Q
    const int half = lane >> 4;
    const int d    = lane & 15;
    unsigned short* dst = (op == 0) ? Q : (op == 1) ? Kd : V;
#pragma unroll
    for (int r = 0; r < 8; ++r) {
        int row = mtile * 16 + r + half * 8;          // == b*T + t
        int b = row / SEQ, t = row % SEQ;
        size_t idx = (((size_t)b * NUM_HEADS + h) * SEQ + t) * HEAD_SIZE + d;
        dst[idx] = f2bf(c[r] * scale);
    }
}

// --------------------------------------------------------- flash attention
// One wave handles 64 query rows (4 WMMA tiles) so each shared K fragment /
// V column-gather feeds 4 score WMMAs + 4 P*V WMMAs.
__global__ void flash_attn_kernel(const unsigned short* __restrict__ Q,
                                  const unsigned short* __restrict__ Kd,
                                  const unsigned short* __restrict__ V,
                                  unsigned short* __restrict__ attnb)
{
    __shared__ unsigned short Plds[MTILES][16][18];
    const int lane = threadIdx.x;
    const int half = lane >> 4;
    const int l15  = lane & 15;
    const int m0 = blockIdx.x * (16 * MTILES);
    const int h  = blockIdx.y;
    const int b  = blockIdx.z;
    const size_t base = (((size_t)b * NUM_HEADS + h) * SEQ) * HEAD_SIZE;
    const unsigned short* Qb = Q  + base;
    const unsigned short* Kb = Kd + base;
    const unsigned short* Vb = V  + base;

    const v8u zu = {0,0,0,0,0,0,0,0};
    const v8f zf = {0.f,0.f,0.f,0.f,0.f,0.f,0.f,0.f};

    // Q A-fragments: 16x16 tiles zero-padded to K=32 (lanes 16-31 = 0)
    v16bf qa[MTILES];
#pragma unroll
    for (int mt = 0; mt < MTILES; ++mt) {
        v8u qu = zu;
        const unsigned short* qrow =
            Qb + (size_t)(m0 + mt * 16 + l15) * HEAD_SIZE;
#pragma unroll
        for (int j = 0; j < 8; ++j)
            if (half == 0) qu[j] = *(const unsigned int*)(qrow + 2*j);
        qa[mt] = __builtin_bit_cast(v16bf, qu);
    }

    float rmax[MTILES][8], rsum[MTILES][8];
    v8f o[MTILES];
#pragma unroll
    for (int mt = 0; mt < MTILES; ++mt) {
        o[mt] = zf;
#pragma unroll
        for (int r = 0; r < 8; ++r) { rmax[mt][r] = -3.0e38f; rsum[mt][r] = 0.f; }
    }

    for (int s0 = 0; s0 <= m0 + 16 * (MTILES - 1); s0 += 16) {
        // K^T B-fragment (shared): lanes 0-15 = K-rows d of column s=s0+l15;
        // lanes 16-31 = zero padding (K rows 16..31).
        v8u ku = zu;
        if (half == 0) {
            const unsigned short* krow = Kb + (size_t)(s0 + l15) * HEAD_SIZE;
#pragma unroll
            for (int j = 0; j < 8; ++j)
                ku[j] = *(const unsigned int*)(krow + 2*j);
        }
        v16bf kb = __builtin_bit_cast(v16bf, ku);
        // V B-fragment (shared): lanes 0-15: column d=l15, K-rows s=s0..s0+15
        v8u vu = zu;
        if (half == 0) {
#pragma unroll
            for (int e = 0; e < 16; e += 2) {
                unsigned int lo = Vb[(size_t)(s0 + e    ) * HEAD_SIZE + l15];
                unsigned int hi = Vb[(size_t)(s0 + e + 1) * HEAD_SIZE + l15];
                vu[e >> 1] = lo | (hi << 16);
            }
        }
        v16bf vb = __builtin_bit_cast(v16bf, vu);

#pragma unroll
        for (int mt = 0; mt < MTILES; ++mt) {
            const int mb = m0 + 16 * mt;
            if (s0 > mb) continue;            // block-uniform: fully masked
            v8f s = __builtin_amdgcn_wmma_f32_16x16x32_bf16(
                false, qa[mt], false, kb, (short)0, zf, false, false);
            if (s0 == mb) {                   // causal mask on diagonal tile
#pragma unroll
                for (int r = 0; r < 8; ++r)
                    if (l15 > r + half*8) s[r] = -3.0e38f;
            }
#pragma unroll
            for (int r = 0; r < 8; ++r) {     // online softmax per row
                float tmax = s[r];
                tmax = fmaxf(tmax, __shfl_xor(tmax, 1, 32));
                tmax = fmaxf(tmax, __shfl_xor(tmax, 2, 32));
                tmax = fmaxf(tmax, __shfl_xor(tmax, 4, 32));
                tmax = fmaxf(tmax, __shfl_xor(tmax, 8, 32));
                float nmax = fmaxf(rmax[mt][r], tmax);
                float corr = __expf(rmax[mt][r] - nmax);
                rmax[mt][r] = nmax;
                float p  = __expf(s[r] - nmax);
                float ps = p;
                ps += __shfl_xor(ps, 1, 32);
                ps += __shfl_xor(ps, 2, 32);
                ps += __shfl_xor(ps, 4, 32);
                ps += __shfl_xor(ps, 8, 32);
                rsum[mt][r] = rsum[mt][r] * corr + ps;
                o[mt][r]   *= corr;
                Plds[mt][r + half*8][l15] = f2bf(p);
            }
        }
        __syncthreads();
#pragma unroll
        for (int mt = 0; mt < MTILES; ++mt) { // C-layout -> A-layout via LDS
            const int mb = m0 + 16 * mt;
            if (s0 > mb) continue;
            v8u pu = zu;
#pragma unroll
            for (int j = 0; j < 8; ++j)
                if (half == 0) pu[j] = *(const unsigned int*)(&Plds[mt][l15][2*j]);
            v16bf pa = __builtin_bit_cast(v16bf, pu);
            o[mt] = __builtin_amdgcn_wmma_f32_16x16x32_bf16(
                false, pa, false, vb, (short)0, o[mt], false, false);
        }
        __syncthreads();
    }

#pragma unroll
    for (int mt = 0; mt < MTILES; ++mt)       // normalize + store [B,T,C] bf16
#pragma unroll
        for (int r = 0; r < 8; ++r) {
            float val = o[mt][r] / rsum[mt][r];
            int t = m0 + mt * 16 + r + half*8;
            size_t idx = ((size_t)b * SEQ + t) * N_EMBD + h * HEAD_SIZE + l15;
            attnb[idx] = f2bf(val);
        }
}

// ------------------------------------------------------- output projection
__global__ void proj_gemm_kernel(const unsigned short* __restrict__ attnb,
                                 const unsigned short* __restrict__ wp,
                                 const float* __restrict__ bias,
                                 float* __restrict__ out)
{
    const int lane  = threadIdx.x;
    const int mtile = blockIdx.x;
    const int nt    = blockIdx.y;
    v8f c = gemm16x16xC(attnb + (size_t)mtile * 16 * N_EMBD, N_EMBD,
                        wp + (size_t)nt * 16 * N_EMBD, N_EMBD, lane);
    const int half = lane >> 4;
    const int n = nt * 16 + (lane & 15);
    const float bv = bias[n];
#pragma unroll
    for (int r = 0; r < 8; ++r) {
        int row = mtile * 16 + r + half * 8;
        out[(size_t)row * N_EMBD + n] = c[r] + bv;
    }
}

// ------------------------------------------------------------------ launch
extern "C" void kernel_launch(void* const* d_in, const int* in_sizes, int n_in,
                              void* d_out, int out_size, void* d_ws, size_t ws_size,
                              hipStream_t stream) {
    const float* x  = (const float*)d_in[0];
    const float* Wk = (const float*)d_in[1];
    const float* Wq = (const float*)d_in[2];
    const float* Wv = (const float*)d_in[3];
    const float* Wp = (const float*)d_in[4];
    const float* bp = (const float*)d_in[5];
    (void)in_sizes; (void)n_in; (void)out_size; (void)ws_size;

    const size_t NX = (size_t)BATCH * SEQ * N_EMBD;    // 1,048,576
    const size_t NW = (size_t)N_EMBD * N_EMBD;         // 16,384
    char* ws = (char*)d_ws;
    size_t off = 0;
    auto take = [&](size_t bytes) -> unsigned short* {
        unsigned short* p = (unsigned short*)(ws + off);
        off += (bytes + 255) & ~(size_t)255;
        return p;
    };
    unsigned short* xb    = take(NX * 2);
    unsigned short* wqb   = take(NW * 2);
    unsigned short* wkb   = take(NW * 2);
    unsigned short* wvb   = take(NW * 2);
    unsigned short* wpb   = take(NW * 2);
    unsigned short* Qb    = take(NX * 2);
    unsigned short* Kb    = take(NX * 2);
    unsigned short* Vb    = take(NX * 2);
    unsigned short* attnb = take(NX * 2);

    cvt_bf16_kernel<<<1024, 256, 0, stream>>>(x,  xb,  (int)NX);
    cvt_bf16_kernel<<<64,   256, 0, stream>>>(Wq, wqb, (int)NW);
    cvt_bf16_kernel<<<64,   256, 0, stream>>>(Wk, wkb, (int)NW);
    cvt_bf16_kernel<<<64,   256, 0, stream>>>(Wv, wvb, (int)NW);
    cvt_bf16_kernel<<<64,   256, 0, stream>>>(Wp, wpb, (int)NW);

    qkv_gemm_kernel<<<dim3(BATCH * SEQ / 16, NUM_HEADS, 3), 32, 0, stream>>>(
        xb, wqb, wkb, wvb, Qb, Kb, Vb);

    flash_attn_kernel<<<dim3(SEQ / (16 * MTILES), NUM_HEADS, BATCH), 32, 0, stream>>>(
        Qb, Kb, Vb, attnb);

    proj_gemm_kernel<<<dim3(BATCH * SEQ / 16, N_EMBD / 16), 32, 0, stream>>>(
        attnb, wpb, bp, (float*)d_out);
}
